// DMFM_9216999817365
// MI455X (gfx1250) — matrix operand order repaired and checked by reading, not compiled, not run on hardware
//
#include <hip/hip_runtime.h>
#include <math.h>

typedef __attribute__((ext_vector_type(16))) _Float16 v16h;
typedef __attribute__((ext_vector_type(8)))  float    v8f;
typedef __attribute__((ext_vector_type(4)))  unsigned int u32x4;
typedef __attribute__((ext_vector_type(8)))  int i32x8;
typedef __attribute__((ext_vector_type(4)))  int i32x4;

constexpr int   NB = 2, NN = 2000, NF = 64, ND = 128, NH = 4, DH = 32, HN = 3, DEEP = 384;
constexpr float SLOPE = 0.1f;

// d_out offsets (in floats), reference tuple order
constexpr size_t OFF_C    = 0;
constexpr size_t SZ_C     = (size_t)NB * NN * ND;        // 512000
constexpr size_t OFF_CBI  = 1 * SZ_C;
constexpr size_t OFF_CBU  = 2 * SZ_C;
constexpr size_t OFF_HI   = 3 * SZ_C;
constexpr size_t OFF_HU   = 4 * SZ_C;
constexpr size_t OFF_FACT = 5 * SZ_C;
constexpr size_t OFF_REC  = OFF_FACT + (size_t)HN * NB * NN;
constexpr size_t OFF_A    = OFF_REC  + (size_t)HN * NB * NN;

#if defined(__has_builtin)
#if __has_builtin(__builtin_amdgcn_tensor_load_to_lds)
#define HAVE_TDM 1
#endif
#endif

__device__ __forceinline__ float lrelu(float x) { return x > 0.f ? x : SLOPE * x; }

__device__ __forceinline__ v8f wmma16(v16h a, v16h b, v8f c) {
  // D = A(16x32 f16) * B(32x16 f16) + C(16x16 f32)
  return __builtin_amdgcn_wmma_f32_16x16x32_f16(false, a, false, b, (short)0, c,
                                                false, false);
}

// A-frag (16x32 f16) from row-major LDS X[16][ld]; lane holds row lane&15,
// K elems: base+e, base=(lane>>4)*16
__device__ __forceinline__ v16h a_frag_lds(const _Float16* X, int ld, int lane, int kk) {
  int r = lane & 15, base = (lane >> 4) << 4;
  v16h a;
#pragma unroll
  for (int e = 0; e < 16; ++e) {
    int k = kk + base + e;
    a[e] = X[r * ld + k];
  }
  return a;
}

// B-frag (32x16) from global f32 W[k][ld]; lane holds col lane&15,
// K rows: (lane>>4)*16 + e
__device__ __forceinline__ v16h b_frag_g32(const float* W, int ld, int lane, int kk, int cc) {
  int n = lane & 15;
  int kr = kk + ((lane >> 4) << 4);
  v16h b;
#pragma unroll
  for (int e = 0; e < 16; ++e) b[e] = (_Float16)W[(size_t)(kr + e) * ld + cc + n];
  return b;
}

#ifdef HAVE_TDM
// TDM: DMA a (rows_valid x DH) f16 tile (tile_dim 32x32, OOB rows -> zero)
// from global into LDS. Descriptor per CDNA5 ISA.
__device__ __forceinline__ void tdm_load_tile(unsigned int lds_off,
                                              const _Float16* gsrc, int rows_valid) {
  unsigned long long ga = (unsigned long long)(size_t)gsrc;
  u32x4 g0;
  g0[0] = 1u;                                     // count=1, user mode
  g0[1] = lds_off;                                // lds_addr [63:32]
  g0[2] = (unsigned int)ga;                       // global_addr [95:64]
  g0[3] = (unsigned int)(ga >> 32) | 0x80000000u; // addr[56:32] | type=2 [127:126]
  i32x8 g1;
  g1[0] = 0x00010000;                      // wg_mask=0, data_size=1 (2 bytes)
  g1[1] = (int)((unsigned)DH << 16);       // tensor_dim0[15:0]=32 @ [63:48]
  g1[2] = (int)((unsigned)rows_valid << 16); // dim0 hi=0 | tensor_dim1[15:0]
  g1[3] = (int)(32u << 16);                // dim1 hi=0 | tile_dim0=32 @ [127:112]
  g1[4] = 32;                              // tile_dim1=32, tile_dim2=0
  g1[5] = DH;                              // tensor_dim0_stride=32 (lo32 of 48)
  g1[6] = 0;                               // stride0 hi | tensor_dim1_stride lo
  g1[7] = 0;
  i32x4 z4 = {};
#if __clang_major__ >= 23
  i32x8 z8 = {};
  __builtin_amdgcn_tensor_load_to_lds(g0, g1, z4, z4, z8, 0);
#else
  __builtin_amdgcn_tensor_load_to_lds(g0, g1, z4, z4, 0);
#endif
}
#endif

// ---------------- encoder: C = relu(BN(x)@W1+b1)@W2+b2 ----------------
__global__ void __launch_bounds__(32)
encoder_kernel(const float* __restrict__ feats, const float* __restrict__ gam,
               const float* __restrict__ bet, const float* __restrict__ mu,
               const float* __restrict__ var, const float* __restrict__ W1,
               const float* __restrict__ b1, const float* __restrict__ W2,
               const float* __restrict__ b2, float* __restrict__ C) {
  __shared__ _Float16 xh[16 * NF];
  __shared__ _Float16 hid[16 * ND];
  int lane = threadIdx.x;
  int idx0 = blockIdx.x * 16;  // flat (b*N+n) row tile
  for (int t = lane; t < 16 * NF; t += 32) {
    int f = t & 63;
    float x = feats[(size_t)idx0 * NF + t];
    xh[t] = (_Float16)((x - mu[f]) * rsqrtf(var[f] + 1e-5f) * gam[f] + bet[f]);
  }
  __syncthreads();
  v16h a0 = a_frag_lds(xh, NF, lane, 0);
  v16h a1 = a_frag_lds(xh, NF, lane, 32);
  int n = lane & 15, rb = (lane >> 4) << 3;
#pragma unroll
  for (int ct = 0; ct < 8; ++ct) {
    v8f c = {};
    c = wmma16(a0, b_frag_g32(W1, ND, lane, 0, ct * 16), c);
    c = wmma16(a1, b_frag_g32(W1, ND, lane, 32, ct * 16), c);
    float bias = b1[ct * 16 + n];
#pragma unroll
    for (int v = 0; v < 8; ++v) {
      float h = c[v] + bias;
      hid[(rb + v) * ND + ct * 16 + n] = (_Float16)(h > 0.f ? h : 0.f);
    }
  }
  __syncthreads();
  v16h af[4];
#pragma unroll
  for (int kt = 0; kt < 4; ++kt) af[kt] = a_frag_lds(hid, ND, lane, kt * 32);
#pragma unroll
  for (int ct = 0; ct < 8; ++ct) {
    v8f c = {};
#pragma unroll
    for (int kt = 0; kt < 4; ++kt)
      c = wmma16(af[kt], b_frag_g32(W2, ND, lane, kt * 32, ct * 16), c);
    float bias = b2[ct * 16 + n];
#pragma unroll
    for (int v = 0; v < 8; ++v)
      C[(size_t)(idx0 + rb + v) * ND + ct * 16 + n] = c[v] + bias;
  }
}

// ---------------- GAT projection: h = X@W, scores s_src/s_dst -------------
__global__ void __launch_bounds__(32)
proj_kernel(const float* __restrict__ X, const float* __restrict__ W,
            const float* __restrict__ a_src, const float* __restrict__ a_dst,
            _Float16* __restrict__ h16, float* __restrict__ ssrc,
            float* __restrict__ sdst) {
  __shared__ _Float16 xh[16 * ND];
  __shared__ float hs[16 * ND];
  int lane = threadIdx.x;
  int idx0 = blockIdx.x * 16;
  int b = idx0 / NN, i0 = idx0 - b * NN;
  for (int t = lane; t < 16 * ND; t += 32)
    xh[t] = (_Float16)X[(size_t)idx0 * ND + t];
  __syncthreads();
  v16h af[4];
#pragma unroll
  for (int kt = 0; kt < 4; ++kt) af[kt] = a_frag_lds(xh, ND, lane, kt * 32);
  int n = lane & 15, rb = (lane >> 4) << 3;
#pragma unroll
  for (int ct = 0; ct < 8; ++ct) {
    v8f c = {};
#pragma unroll
    for (int kt = 0; kt < 4; ++kt)
      c = wmma16(af[kt], b_frag_g32(W, ND, lane, kt * 32, ct * 16), c);
#pragma unroll
    for (int v = 0; v < 8; ++v) hs[(rb + v) * ND + ct * 16 + n] = c[v];
  }
  __syncthreads();
  // h16: (B, NH, N, DH) f16 for attn B-tiles (TDM source)
  for (int t = lane; t < 16 * ND; t += 32) {
    int r = t >> 7, d = t & 127;
    int hd = d >> 5, dd = d & 31;
    h16[(((size_t)b * NH + hd) * NN + (i0 + r)) * DH + dd] = (_Float16)hs[t];
  }
  // per-(row,head) attention score halves
  for (int p = lane; p < 64; p += 32) {
    int r = p >> 2, hd = p & 3;
    float as = 0.f, ad = 0.f;
    for (int d = 0; d < DH; ++d) {
      float hv = hs[r * ND + hd * DH + d];
      as += hv * a_src[hd * DH + d];
      ad += hv * a_dst[hd * DH + d];
    }
    size_t o = ((size_t)b * NH + hd) * NN + (i0 + r);
    ssrc[o] = as; sdst[o] = ad;
  }
}

// ---------------- masked softmax row stats (online LSE, all 4 heads) ------
__global__ void __launch_bounds__(128)
rowstats_kernel(const unsigned char* __restrict__ mask, const float* __restrict__ ssrc,
                const float* __restrict__ sdst, float* __restrict__ rmax,
                float* __restrict__ rinv) {
  int t = blockIdx.x * 128 + threadIdx.x;
  if (t >= NB * NN) return;
  int b = t / NN, i = t - b * NN;
  const unsigned char* mrow = mask + (size_t)t * NN;
  const float* sd0 = sdst + ((size_t)b * NH + 0) * NN;
  const float* sd1 = sdst + ((size_t)b * NH + 1) * NN;
  const float* sd2 = sdst + ((size_t)b * NH + 2) * NN;
  const float* sd3 = sdst + ((size_t)b * NH + 3) * NN;
  float ss[NH], m[NH], l[NH];
#pragma unroll
  for (int h = 0; h < NH; ++h) {
    ss[h] = ssrc[((size_t)b * NH + h) * NN + i];
    m[h] = -1e30f; l[h] = 0.f;
  }
  for (int j0 = 0; j0 < NN; j0 += 8) {
    unsigned long long m64 = *(const unsigned long long*)(mrow + j0);
    if (!m64) continue;
#pragma unroll
    for (int e = 0; e < 8; ++e) {
      if ((m64 >> (8 * e)) & 0xff) {
        int j = j0 + e;
        float sj[NH] = {sd0[j], sd1[j], sd2[j], sd3[j]};
#pragma unroll
        for (int h = 0; h < NH; ++h) {
          float s = lrelu(ss[h] + sj[h]);
          if (s > m[h]) { l[h] = l[h] * __expf(m[h] - s) + 1.f; m[h] = s; }
          else          { l[h] += __expf(s - m[h]); }
        }
      }
    }
  }
#pragma unroll
  for (int h = 0; h < NH; ++h) {
    size_t o = ((size_t)b * NH + h) * NN + i;
    rmax[o] = m[h]; rinv[o] = 1.f / l[h];
  }
}

// ---------------- attention aggregation: H = softmax(e) @ h  (WMMA+TDM) ---
// also writes Cbar = Xres - H  (residual subtraction)
__global__ void __launch_bounds__(32)
attn_kernel(const unsigned char* __restrict__ mask, const _Float16* __restrict__ h16,
            const float* __restrict__ ssrc, const float* __restrict__ sdst,
            const float* __restrict__ rmax, const float* __restrict__ rinv,
            const float* __restrict__ Xres, float* __restrict__ Hout,
            float* __restrict__ Cbar) {
  __shared__ _Float16 hbuf[2][32 * DH];  // double-buffered 32x32 f16 h tile
  int lane = threadIdx.x;
  int rt = blockIdx.x, hd = blockIdx.y, b = blockIdx.z;
  int i0 = rt * 16;
  int r = lane & 15, base = (lane >> 4) << 3;
  int i = i0 + r;
  size_t bh = (size_t)b * NH + hd;
  float s_i  = ssrc[bh * NN + i];
  float m_i  = rmax[bh * NN + i];
  float il_i = rinv[bh * NN + i];
  const unsigned char* mrow = mask + ((size_t)b * NN + i) * NN;
  const float* sd = sdst + bh * NN;
  const _Float16* hb = h16 + bh * NN * DH;
  int nn = lane & 15, koff = (lane >> 4) << 4;
  v8f acc0 = {}, acc1 = {};

#ifdef HAVE_TDM
  unsigned int ldsA = (unsigned int)(size_t)(void*)&hbuf[0][0];
  unsigned int ldsB = (unsigned int)(size_t)(void*)&hbuf[1][0];
  tdm_load_tile(ldsA, hb, NN);  // prologue: tile 0
#endif
  int buf = 0;
  for (int kk = 0; kk < NN; kk += 32) {   // N=2000 -> last tile partial (16)
    const bool hasNext = (kk + 32) < NN;
#ifdef HAVE_TDM
    if (hasNext) {
      asm volatile("s_wait_dscnt 0x0" ::: "memory");  // LDS reads done before DMA reuse
      tdm_load_tile(buf ? ldsA : ldsB, hb + (size_t)(kk + 32) * DH, NN - (kk + 32));
    }
#else
    {
      // fallback staging: lane == row of the 32x32 tile
      const _Float16* src = hb + (size_t)(kk + lane) * DH;
      _Float16* dst = &hbuf[buf][lane * DH];
      if (kk + lane < NN) for (int q = 0; q < DH; ++q) dst[q] = src[q];
      else                for (int q = 0; q < DH; ++q) dst[q] = (_Float16)0.f;
      __syncthreads();
    }
#endif
    // ---- A-frag: masked, normalized probabilities (two 8-wide K runs) ----
    unsigned long long mA = *(const unsigned long long*)(mrow + kk + base);
    v8f sA = *(const v8f*)(sd + kk + base);
    bool v2 = (kk + 16 + base + 8) <= NN;   // tail chunk is all-or-nothing
    unsigned long long mB = v2 ? *(const unsigned long long*)(mrow + kk + 16 + base) : 0ull;
    v8f sB = v2 ? *(const v8f*)(sd + kk + 16 + base) : (v8f){};
    v16h a;
#pragma unroll
    for (int e = 0; e < 8; ++e) {
      float p = 0.f;
      if ((mA >> (8 * e)) & 0xff) {
        float s = lrelu(s_i + sA[e]);
        p = __expf(s - m_i) * il_i;
      }
      a[e] = (_Float16)p;
    }
#pragma unroll
    for (int e = 0; e < 8; ++e) {
      float p = 0.f;
      if ((mB >> (8 * e)) & 0xff) {
        float s = lrelu(s_i + sB[e]);
        p = __expf(s - m_i) * il_i;
      }
      a[8 + e] = (_Float16)p;
    }
#ifdef HAVE_TDM
    if (hasNext) asm volatile("s_wait_tensorcnt 0x1" ::: "memory");
    else         asm volatile("s_wait_tensorcnt 0x0" ::: "memory");
#endif
    // ---- B-frags from LDS tile ----
    const _Float16* ht = &hbuf[buf][0];
    v16h b0, b1;
#pragma unroll
    for (int e = 0; e < 16; ++e) {
      b0[e] = ht[(e < 8 ? base + e : 8 + base + e) * DH + nn];
      b1[e] = ht[(e < 8 ? base + e : 8 + base + e) * DH + 16 + nn];
    }
    acc0 = wmma16(a, b0, acc0);
    acc1 = wmma16(a, b1, acc1);
    buf ^= 1;
  }
#pragma unroll
  for (int v = 0; v < 8; ++v) {
    int ig = i0 + base + v;
    size_t o0 = ((size_t)b * NN + ig) * ND + hd * DH + nn;
    size_t o1 = o0 + 16;
    float h0 = acc0[v], h1 = acc1[v];
    Hout[o0] = h0; Cbar[o0] = Xres[o0] - h0;
    Hout[o1] = h1; Cbar[o1] = Xres[o1] - h1;
  }
}

// ---------------- per-horizon deep-factor heads ----------------
__global__ void __launch_bounds__(256)
factors_kernel(const float* __restrict__ Cc, const float* __restrict__ Ci,
               const float* __restrict__ Cu, const float* __restrict__ hW,
               const float* __restrict__ hbias, float* __restrict__ out) {
  int t = blockIdx.x * 256 + threadIdx.x;
  if (t >= HN * NB * NN) return;
  int hbidx = t / NN, n = t - hbidx * NN;
  int hh = hbidx / NB, b = hbidx - hh * NB;
  const float* w = hW + (size_t)hh * DEEP;
  size_t ro = ((size_t)b * NN + n) * ND;
  float s = hbias[hh];
  for (int d = 0; d < ND; ++d) s += Cc[ro + d] * w[d];
  for (int d = 0; d < ND; ++d) s += Ci[ro + d] * w[ND + d];
  for (int d = 0; d < ND; ++d) s += Cu[ro + d] * w[2 * ND + d];
  out[t] = lrelu(s);
}

// ---------------- factor attention: A = softmax(leaky(x @ P^T)) (WMMA) ----
__global__ void __launch_bounds__(32)
fattn_kernel(const float* __restrict__ feats, const float* __restrict__ pW,
             float* __restrict__ A) {
  __shared__ _Float16 xh[16 * NF];
  __shared__ float uf[16 * NF];
  int lane = threadIdx.x;
  int rt = blockIdx.x, b = blockIdx.y, hh = blockIdx.z;
  int n0 = rt * 16;
  for (int t = lane; t < 16 * NF; t += 32)
    xh[t] = (_Float16)feats[((size_t)b * NN + n0) * NF + t];
  __syncthreads();
  v16h a0 = a_frag_lds(xh, NF, lane, 0);
  v16h a1 = a_frag_lds(xh, NF, lane, 32);
  int n = lane & 15, rb = (lane >> 4) << 3, koff = (lane >> 4) << 4;
  const float* Wh = pW + (size_t)hh * NF * NF;  // [k_out][m]: B = Wh^T view
#pragma unroll
  for (int ct = 0; ct < 4; ++ct) {
    int kc = ct * 16 + n;
    v16h b0, b1;
#pragma unroll
    for (int e = 0; e < 16; ++e) {
      b0[e] = (_Float16)Wh[(size_t)kc * NF + koff + e];
      b1[e] = (_Float16)Wh[(size_t)kc * NF + 32 + koff + e];
    }
    v8f c = {};
    c = wmma16(a0, b0, c);
    c = wmma16(a1, b1, c);
#pragma unroll
    for (int v = 0; v < 8; ++v) uf[(rb + v) * NF + kc] = lrelu(c[v]);
  }
  __syncthreads();
  if (lane < 16) {
    float mx = -1e30f;
    for (int k = 0; k < NF; ++k) mx = fmaxf(mx, uf[lane * NF + k]);
    float sum = 0.f;
    for (int k = 0; k < NF; ++k) sum += __expf(uf[lane * NF + k] - mx);
    float inv = 1.f / sum;
    size_t ao = (((size_t)hh * NB + b) * NN + (n0 + lane)) * NF;
    for (int k = 0; k < NF; ++k) A[ao + k] = __expf(uf[lane * NF + k] - mx) * inv;
  }
}

// ---------------- a_bar = mean_n A ; recon = feats @ a_bar ----------------
__global__ void __launch_bounds__(64)
abar_kernel(const float* __restrict__ A, float* __restrict__ abar) {
  int hb = blockIdx.x, m = threadIdx.x;
  const float* Ab = A + (size_t)hb * NN * NF;
  float s = 0.f;
  for (int nIdx = 0; nIdx < NN; ++nIdx) s += Ab[(size_t)nIdx * NF + m];
  abar[hb * NF + m] = s * (1.f / NN);
}

__global__ void __launch_bounds__(256)
recon_kernel(const float* __restrict__ feats, const float* __restrict__ abar,
             float* __restrict__ out) {
  int t = blockIdx.x * 256 + threadIdx.x;
  if (t >= HN * NB * NN) return;
  int hb = t / NN, n = t - hb * NN;
  int b = hb % NB;
  const float* fr = feats + ((size_t)b * NN + n) * NF;
  const float* ab = abar + (size_t)hb * NF;
  float s = 0.f;
  for (int m = 0; m < NF; ++m) s += fr[m] * ab[m];
  out[t] = s;
}

extern "C" void kernel_launch(void* const* d_in, const int* in_sizes, int n_in,
                              void* d_out, int out_size, void* d_ws, size_t ws_size,
                              hipStream_t stream) {
  (void)in_sizes; (void)n_in; (void)out_size; (void)ws_size;
  const float*         feats   = (const float*)d_in[0];
  const unsigned char* indm    = (const unsigned char*)d_in[1];
  const unsigned char* unim    = (const unsigned char*)d_in[2];
  const float* bn_gamma = (const float*)d_in[3];
  const float* bn_beta  = (const float*)d_in[4];
  const float* bn_mean  = (const float*)d_in[5];
  const float* bn_var   = (const float*)d_in[6];
  const float* W1 = (const float*)d_in[7];
  const float* b1 = (const float*)d_in[8];
  const float* W2 = (const float*)d_in[9];
  const float* b2 = (const float*)d_in[10];
  const float* giW = (const float*)d_in[11];
  const float* gia_src = (const float*)d_in[12];
  const float* gia_dst = (const float*)d_in[13];
  const float* guW = (const float*)d_in[14];
  const float* gua_src = (const float*)d_in[15];
  const float* gua_dst = (const float*)d_in[16];
  const float* head_W = (const float*)d_in[17];
  const float* head_b = (const float*)d_in[18];
  const float* proj_W = (const float*)d_in[19];
  float* out = (float*)d_out;

  // workspace layout (bytes)
  char* ws = (char*)d_ws;
  _Float16* h16 = (_Float16*)ws;                              // B*NH*N*DH f16 = 1,024,000 B
  float* ssrc = (float*)(ws + 1024000);                       // B*NH*N f32
  float* sdst = (float*)(ws + 1088000);
  float* rmax = (float*)(ws + 1152000);
  float* rinv = (float*)(ws + 1216000);
  float* abar = (float*)(ws + 1280000);                       // HN*NB*NF f32

  const int rowTiles = NB * NN / 16;        // 250
  const int rsBlocks = (NB * NN + 127) / 128;

  encoder_kernel<<<rowTiles, 32, 0, stream>>>(feats, bn_gamma, bn_beta, bn_mean,
                                              bn_var, W1, b1, W2, b2, out + OFF_C);
  // GAT 1 (industry)
  proj_kernel<<<rowTiles, 32, 0, stream>>>(out + OFF_C, giW, gia_src, gia_dst,
                                           h16, ssrc, sdst);
  rowstats_kernel<<<rsBlocks, 128, 0, stream>>>(indm, ssrc, sdst, rmax, rinv);
  attn_kernel<<<dim3(NN / 16, NH, NB), 32, 0, stream>>>(indm, h16, ssrc, sdst, rmax,
                                                        rinv, out + OFF_C,
                                                        out + OFF_HI, out + OFF_CBI);
  // GAT 2 (universe)
  proj_kernel<<<rowTiles, 32, 0, stream>>>(out + OFF_CBI, guW, gua_src, gua_dst,
                                           h16, ssrc, sdst);
  rowstats_kernel<<<rsBlocks, 128, 0, stream>>>(unim, ssrc, sdst, rmax, rinv);
  attn_kernel<<<dim3(NN / 16, NH, NB), 32, 0, stream>>>(unim, h16, ssrc, sdst, rmax,
                                                        rinv, out + OFF_CBI,
                                                        out + OFF_HU, out + OFF_CBU);
  // heads + factor attention
  factors_kernel<<<(HN * NB * NN + 255) / 256, 256, 0, stream>>>(
      out + OFF_C, out + OFF_CBI, out + OFF_CBU, head_W, head_b, out + OFF_FACT);
  fattn_kernel<<<dim3(NN / 16, NB, HN), 32, 0, stream>>>(feats, proj_W, out + OFF_A);
  abar_kernel<<<HN * NB, 64, 0, stream>>>(out + OFF_A, abar);
  recon_kernel<<<(HN * NB * NN + 255) / 256, 256, 0, stream>>>(feats, abar,
                                                               out + OFF_REC);
}
